// DynamicConv2d_40553081209198
// MI455X (gfx1250) — compile-verified, matrix operand order checked by reading, not run on hardware
//
#include <hip/hip_runtime.h>
#include <math.h>

// ---------------- problem constants ----------------
#define BB 32
#define CC 256
#define HH 56
#define WW 56
#define OC 256
#define MM 8
#define HP 58            // padded H (56 + 2)
#define WP 58            // padded W
#define KDIM 2304        // C * 9, ordered (kh,kw)-major, ic-minor
#define NPIX 3136        // 56*56

typedef __attribute__((ext_vector_type(16))) __bf16 v16bf;
typedef __attribute__((ext_vector_type(8)))  float  v8f;

// workspace layout (bytes)
#define X2_ELEMS   ((size_t)BB * HP * WP * CC)            // 27,557,888 bf16
#define X2_BYTES   (X2_ELEMS * 2)                          // 55,115,776
#define WC_ELEMS   ((size_t)BB * OC * KDIM)                // 18,874,368 bf16
#define WC_BYTES   (WC_ELEMS * 2)                          // 37,748,736

// ---------------------------------------------------------------------------
// K1: NCHW fp32 -> padded NHWC bf16 (interior only; halo zeroed by memset)
// grid (B, H), 256 threads. LDS-staged so global reads are w-contiguous.
// ---------------------------------------------------------------------------
__global__ __launch_bounds__(256)
void k1_convert_pad(const float* __restrict__ x, __bf16* __restrict__ x2) {
  const int b = blockIdx.x;
  const int h = blockIdx.y;
  __shared__ float tile[4 * WW];
  const int t = threadIdx.x;
  for (int ic0 = 0; ic0 < CC; ic0 += 4) {
    if (t < 4 * WW) {
      const int icl = t / WW;
      const int w   = t - icl * WW;
      tile[icl * WW + w] =
          x[(((size_t)b * CC + ic0 + icl) * HH + h) * WW + w];
    }
    __syncthreads();
    if (t < 4 * WW) {
      const int w2 = t >> 2;         // 0..55
      const int j  = t & 3;          // 0..3
      x2[(((size_t)b * HP + (h + 1)) * WP + (w2 + 1)) * CC + ic0 + j] =
          (__bf16)tile[j * WW + w2];
    }
    __syncthreads();
  }
}

// ---------------------------------------------------------------------------
// K2: per-(b,c) sum over H*W  -> gap  (mean applied later). Deterministic.
// grid (C, B), 256 threads
// ---------------------------------------------------------------------------
__global__ __launch_bounds__(256)
void k2_gap(const float* __restrict__ x, float* __restrict__ gap) {
  const int c = blockIdx.x, b = blockIdx.y;
  __shared__ float red[256];
  const float* xp = x + ((size_t)b * CC + c) * NPIX;
  float s = 0.f;
  for (int p = threadIdx.x; p < NPIX; p += 256) s += xp[p];
  red[threadIdx.x] = s;
  __syncthreads();
  for (int st = 128; st > 0; st >>= 1) {
    if (threadIdx.x < st) red[threadIdx.x] += red[threadIdx.x + st];
    __syncthreads();
  }
  if (threadIdx.x == 0) gap[b * CC + c] = red[0];
}

// ---------------------------------------------------------------------------
// K3: gate logits + softmax -> alpha (B x M). 1 block, 256 threads (b*8+m)
// ---------------------------------------------------------------------------
__global__ __launch_bounds__(256)
void k3_gate(const float* __restrict__ gap, const float* __restrict__ gate_w,
             const float* __restrict__ gate_b, float* __restrict__ alpha) {
  __shared__ float lg[BB * MM];
  const int t = threadIdx.x;
  const int b = t >> 3, m = t & 7;
  const float inv = 1.0f / (float)NPIX;
  float acc = gate_b[m];
  for (int c = 0; c < CC; ++c)
    acc += (gap[b * CC + c] * inv) * gate_w[m * CC + c];
  lg[t] = acc;
  __syncthreads();
  float mx = lg[b * 8];
  #pragma unroll
  for (int i = 1; i < 8; ++i) mx = fmaxf(mx, lg[b * 8 + i]);
  float s = 0.f;
  #pragma unroll
  for (int i = 0; i < 8; ++i) s += __expf(lg[b * 8 + i] - mx);
  alpha[t] = __expf(acc - mx) / s;
}

// ---------------------------------------------------------------------------
// K4: mix expert kernels -> bf16 wcomb[b][oc][khw*256+ic]; mix biases.
// grid (OC, 9), 256 threads (= ic). Each weight element read exactly once.
// ---------------------------------------------------------------------------
__global__ __launch_bounds__(256)
void k4_combine(const float* __restrict__ weight, const float* __restrict__ bias,
                const float* __restrict__ alpha, __bf16* __restrict__ wcomb,
                float* __restrict__ bcomb) {
  const int oc = blockIdx.x, khw = blockIdx.y;
  const int ic = threadIdx.x;
  __shared__ float al[BB * MM];
  al[ic] = alpha[ic];
  __syncthreads();
  float wm[MM];
  #pragma unroll
  for (int m = 0; m < MM; ++m)
    wm[m] = weight[(((size_t)m * OC + oc) * CC + ic) * 9 + khw];
  for (int b = 0; b < BB; ++b) {
    float a = 0.f;
    #pragma unroll
    for (int m = 0; m < MM; ++m) a += wm[m] * al[b * 8 + m];
    wcomb[((size_t)(b * OC + oc)) * KDIM + khw * CC + ic] = (__bf16)a;
  }
  if (khw == 0 && ic < BB) {
    const int b = ic;
    float a = 0.f;
    #pragma unroll
    for (int m = 0; m < MM; ++m) a += al[b * 8 + m] * bias[m * OC + oc];
    bcomb[b * OC + oc] = a;
  }
}

// ---------------------------------------------------------------------------
// K5: implicit-GEMM conv via v_wmma_f32_16x16x32_bf16.
// grid (49 pixTiles, 1, B); 256 threads = 8 waves.
// Block tile: 256 oc x 64 pix. Wave tile: 64 oc x 32 pix
//   (4 A-frags x 2 B-frags -> 8 f32 accumulators, 8 WMMA per K-chunk).
// K loop: 9 taps x 8 chunks of 32 input channels = 72 chunks.
// ---------------------------------------------------------------------------
__global__ __launch_bounds__(256)
void k5_conv_wmma(const __bf16* __restrict__ x2, const __bf16* __restrict__ wcomb,
                  const float* __restrict__ bcomb, float* __restrict__ out) {
  const int b     = blockIdx.z;
  const int pxBlk = blockIdx.x * 64;
  const int tid   = threadIdx.x;
  const int lane  = tid & 31;
  const int wv    = tid >> 5;
  const int ocW   = (wv >> 1) * 64;           // wave's 64 output channels
  const int pxW   = pxBlk + (wv & 1) * 32;    // wave's 32 pixels

  const int l15  = lane & 15;
  const int ksub = (lane >> 4) * 16;          // half-wave K split (0 / 16)

  // A-matrix (mixed kernel) per-lane row pointers: 16 contiguous bf16 per load
  const __bf16* pA0 = wcomb + ((size_t)(b * OC + ocW + l15) * KDIM) + ksub;
  const __bf16* pA1 = pA0 + (size_t)16 * KDIM;
  const __bf16* pA2 = pA0 + (size_t)32 * KDIM;
  const __bf16* pA3 = pA0 + (size_t)48 * KDIM;

  // B-matrix (im2col of padded NHWC x2): per-lane pixel base pointers
  const int p0 = pxW + l15, p1 = p0 + 16;
  const int h0 = p0 / WW, w0 = p0 - h0 * WW;
  const int h1 = p1 / WW, w1 = p1 - h1 * WW;
  const __bf16* pB0b = x2 + ((size_t)(b * HP + h0) * WP + w0) * CC + ksub;
  const __bf16* pB1b = x2 + ((size_t)(b * HP + h1) * WP + w1) * CC + ksub;

  v8f c00 = {}, c01 = {}, c10 = {}, c11 = {};
  v8f c20 = {}, c21 = {}, c30 = {}, c31 = {};

  for (int khw = 0; khw < 9; ++khw) {
    const int dh = khw / 3, dw = khw - 3 * (khw / 3);
    const __bf16* qA0 = pA0 + (size_t)khw * CC;
    const __bf16* qA1 = pA1 + (size_t)khw * CC;
    const __bf16* qA2 = pA2 + (size_t)khw * CC;
    const __bf16* qA3 = pA3 + (size_t)khw * CC;
    const __bf16* qB0 = pB0b + ((size_t)dh * WP + dw) * CC;
    const __bf16* qB1 = pB1b + ((size_t)dh * WP + dw) * CC;
    if (khw < 8) {  // pull next tap's tiles toward L0/L2
      __builtin_prefetch(qA0 + CC, 0, 0);
      __builtin_prefetch(qB0 + ((dw == 2) ? (size_t)WP * CC - 2 * CC : (size_t)CC), 0, 0);
    }
    #pragma unroll
    for (int ci = 0; ci < 8; ++ci) {          // 8 chunks of 32 input channels
      v16bf a0  = *(const v16bf*)(qA0 + ci * 32);
      v16bf a1  = *(const v16bf*)(qA1 + ci * 32);
      v16bf a2  = *(const v16bf*)(qA2 + ci * 32);
      v16bf a3  = *(const v16bf*)(qA3 + ci * 32);
      v16bf bf0 = *(const v16bf*)(qB0 + ci * 32);
      v16bf bf1 = *(const v16bf*)(qB1 + ci * 32);
      c00 = __builtin_amdgcn_wmma_f32_16x16x32_bf16(false, a0, false, bf0,
                                                    (short)0, c00, false, false);
      c01 = __builtin_amdgcn_wmma_f32_16x16x32_bf16(false, a0, false, bf1,
                                                    (short)0, c01, false, false);
      c10 = __builtin_amdgcn_wmma_f32_16x16x32_bf16(false, a1, false, bf0,
                                                    (short)0, c10, false, false);
      c11 = __builtin_amdgcn_wmma_f32_16x16x32_bf16(false, a1, false, bf1,
                                                    (short)0, c11, false, false);
      c20 = __builtin_amdgcn_wmma_f32_16x16x32_bf16(false, a2, false, bf0,
                                                    (short)0, c20, false, false);
      c21 = __builtin_amdgcn_wmma_f32_16x16x32_bf16(false, a2, false, bf1,
                                                    (short)0, c21, false, false);
      c30 = __builtin_amdgcn_wmma_f32_16x16x32_bf16(false, a3, false, bf0,
                                                    (short)0, c30, false, false);
      c31 = __builtin_amdgcn_wmma_f32_16x16x32_bf16(false, a3, false, bf1,
                                                    (short)0, c31, false, false);
    }
  }

  // Epilogue: C/D layout — VGPR r: lanes 0-15 hold M=r, lanes 16-31 hold M=r+8
  const int rowHi = (lane >> 4) * 8;
  const float* bb = bcomb + b * OC;
  float* ob = out + (size_t)b * OC * NPIX;
  const int pc0 = pxW + l15, pc1 = pc0 + 16;
  #pragma unroll
  for (int r = 0; r < 8; ++r) {
    const int o0 = ocW + rowHi + r;
    const int o1 = o0 + 16;
    const int o2 = o0 + 32;
    const int o3 = o0 + 48;
    const float bv0 = bb[o0], bv1 = bb[o1], bv2 = bb[o2], bv3 = bb[o3];
    ob[(size_t)o0 * NPIX + pc0] = c00[r] + bv0;
    ob[(size_t)o0 * NPIX + pc1] = c01[r] + bv0;
    ob[(size_t)o1 * NPIX + pc0] = c10[r] + bv1;
    ob[(size_t)o1 * NPIX + pc1] = c11[r] + bv1;
    ob[(size_t)o2 * NPIX + pc0] = c20[r] + bv2;
    ob[(size_t)o2 * NPIX + pc1] = c21[r] + bv2;
    ob[(size_t)o3 * NPIX + pc0] = c30[r] + bv3;
    ob[(size_t)o3 * NPIX + pc1] = c31[r] + bv3;
  }
}

// ---------------------------------------------------------------------------
extern "C" void kernel_launch(void* const* d_in, const int* in_sizes, int n_in,
                              void* d_out, int out_size, void* d_ws, size_t ws_size,
                              hipStream_t stream) {
  const float* x      = (const float*)d_in[0];  // (B,C,H,W)
  const float* weight = (const float*)d_in[1];  // (M,O,I,3,3)
  const float* bias   = (const float*)d_in[2];  // (M,O)
  const float* gate_w = (const float*)d_in[3];  // (M,C)
  const float* gate_b = (const float*)d_in[4];  // (M,)
  float* out = (float*)d_out;                   // (B,O,H,W)

  char* ws = (char*)d_ws;
  __bf16* x2    = (__bf16*)ws;                          // padded NHWC bf16
  __bf16* wcomb = (__bf16*)(ws + X2_BYTES);             // mixed kernels bf16
  float*  gap   = (float*)(ws + X2_BYTES + WC_BYTES);   // B*C
  float*  alpha = gap + BB * CC;                        // B*M
  float*  bcomb = alpha + BB * MM;                      // B*O

  // zero the halo of x2 (interior is fully overwritten by k1)
  hipMemsetAsync(x2, 0, X2_BYTES, stream);

  k1_convert_pad<<<dim3(BB, HH), 256, 0, stream>>>(x, x2);
  k2_gap        <<<dim3(CC, BB), 256, 0, stream>>>(x, gap);
  k3_gate       <<<1, 256, 0, stream>>>(gap, gate_w, gate_b, alpha);
  k4_combine    <<<dim3(OC, 9), 256, 0, stream>>>(weight, bias, alpha, wcomb, bcomb);
  k5_conv_wmma  <<<dim3(NPIX / 64, 1, BB), 256, 0, stream>>>(x2, wcomb, bcomb, out);
}
